// SlidingWindowAttention_61924838474221
// MI455X (gfx1250) — matrix operand order repaired and checked by reading, not compiled, not run on hardware
//
#include <hip/hip_runtime.h>
#include <hip/hip_bf16.h>

// ---------------------------------------------------------------------------
// Sliding-window attention, MI455X (gfx1250), wave32 + WMMA bf16.
//   B=2, S=4096, D=1024, H=16, dk=64, WINDOW=256, nW=16
// Pipeline:
//   1) f32 -> bf16 convert (vectorized) of q,k,v and wq,wk,wv,wo
//   2) 3x GEMM (X[8192x1024] @ W^T + b) -> Q/K/V bf16
//      B operand staged via async global->LDS (double buffered) when available
//   3) per-window flash attention (DPP row reductions) -> ctx bf16
//   4) GEMM ctx @ wo^T + bo -> f32 d_out
// ---------------------------------------------------------------------------

typedef __attribute__((ext_vector_type(16))) __bf16 v16bf;
typedef __attribute__((ext_vector_type(8)))  __bf16 bf16x8;
typedef __attribute__((ext_vector_type(4)))  __bf16 bf16x4;
typedef __attribute__((ext_vector_type(8)))  float  v8f;
typedef __attribute__((ext_vector_type(4)))  float  f32x4;
typedef __attribute__((ext_vector_type(4)))  int    v4i;

// ---- feature probes --------------------------------------------------------
#if defined(__has_builtin)
#  if __has_builtin(__builtin_amdgcn_global_load_async_to_lds_b128) && \
      __has_builtin(__builtin_amdgcn_s_wait_asynccnt)
#    define SWA_ASYNC 1
#  endif
#  if __has_builtin(__builtin_amdgcn_update_dpp)
#    define SWA_DPP 1
#  endif
#endif
#ifndef SWA_ASYNC
#  define SWA_ASYNC 0
#endif
#ifndef SWA_DPP
#  define SWA_DPP 0
#endif

__device__ __forceinline__ v8f v8f_zero() {
  v8f z;
#pragma unroll
  for (int i = 0; i < 8; ++i) z[i] = 0.0f;
  return z;
}

__device__ __forceinline__ v16bf cat8(bf16x8 lo, bf16x8 hi) {
  return __builtin_shufflevector(lo, hi, 0,1,2,3,4,5,6,7,8,9,10,11,12,13,14,15);
}

__device__ __forceinline__ v16bf ld_ab(const __bf16* p0, const __bf16* p1) {
  return cat8(*(const bf16x8*)p0, *(const bf16x8*)p1);
}

// ---- 16-lane reductions (VALU/DPP16; stays within each 16-lane half) -------
#if SWA_DPP
__device__ __forceinline__ float dpp_mov(float x, int ctrl) {
  int i = __builtin_bit_cast(int, x);
  int r;
  switch (ctrl) {  // ctrl must be an immediate per call site
    case 0: r = __builtin_amdgcn_update_dpp(0, i, 0xB1,  0xf, 0xf, true); break; // quad_perm xor1
    case 1: r = __builtin_amdgcn_update_dpp(0, i, 0x4E,  0xf, 0xf, true); break; // quad_perm xor2
    case 2: r = __builtin_amdgcn_update_dpp(0, i, 0x141, 0xf, 0xf, true); break; // row_half_mirror
    default:r = __builtin_amdgcn_update_dpp(0, i, 0x140, 0xf, 0xf, true); break; // row_mirror
  }
  return __builtin_bit_cast(float, r);
}
__device__ __forceinline__ float red16_max(float x) {
  x = fmaxf(x, dpp_mov(x, 0));
  x = fmaxf(x, dpp_mov(x, 1));
  x = fmaxf(x, dpp_mov(x, 2));
  x = fmaxf(x, dpp_mov(x, 3));
  return x;
}
__device__ __forceinline__ float red16_sum(float x) {
  x += dpp_mov(x, 0);
  x += dpp_mov(x, 1);
  x += dpp_mov(x, 2);
  x += dpp_mov(x, 3);
  return x;
}
#else
__device__ __forceinline__ float red16_max(float x) {
  x = fmaxf(x, __shfl_xor(x, 1, 32));
  x = fmaxf(x, __shfl_xor(x, 2, 32));
  x = fmaxf(x, __shfl_xor(x, 4, 32));
  x = fmaxf(x, __shfl_xor(x, 8, 32));
  return x;
}
__device__ __forceinline__ float red16_sum(float x) {
  x += __shfl_xor(x, 1, 32);
  x += __shfl_xor(x, 2, 32);
  x += __shfl_xor(x, 4, 32);
  x += __shfl_xor(x, 8, 32);
  return x;
}
#endif

#if SWA_ASYNC
__device__ __forceinline__ void async_b128(const __bf16* g, __bf16* l) {
  __builtin_amdgcn_global_load_async_to_lds_b128(
      (__attribute__((address_space(1))) v4i*)g,
      (__attribute__((address_space(3))) v4i*)l, 0, 0);
}
#endif

// ------------------------------ convert ------------------------------------
__global__ __launch_bounds__(256) void swa_cvt_bf16(const float* __restrict__ in,
                                                    __bf16* __restrict__ out, int n4) {
  int i = blockIdx.x * 256 + threadIdx.x;
  const int stride = gridDim.x * 256;
  const f32x4* in4 = (const f32x4*)in;
  bf16x4* out4 = (bf16x4*)out;
  for (; i < n4; i += stride) {
    f32x4 x = in4[i];
    bf16x4 y;
#pragma unroll
    for (int j = 0; j < 4; ++j) y[j] = (__bf16)x[j];
    out4[i] = y;
  }
}

// ------------------------------ GEMM ---------------------------------------
// Y[M,N] = X[M,K] @ W[N,K]^T + bias[N]
// 256 threads = 8 waves; block tile 128x128; wave tile 32x64 (2x4 accums).
// A (16x32 bf16): lane<16 row=l15, K={ak..ak+7,16+ak..}; ak=(lane>>4)*8
// B (32x16 bf16): lane col=l15, K={bk..bk+7,16+bk..}; bk=(lane>>4)*8  (B col n = W row n)
template <bool OUT_F32>
__global__ __launch_bounds__(256) void swa_gemm_bf16(
    const __bf16* __restrict__ X, const __bf16* __restrict__ W,
    const float* __restrict__ bias, void* __restrict__ Yv,
    int M, int N, int K) {
  const int lane = threadIdx.x & 31;
  const int wid  = threadIdx.x >> 5;
  const int wm = wid & 3, wn = wid >> 2;   // 4 waves along M, 2 along N
  const int l15 = lane & 15, lhi = lane >> 4;

  const int arow0 = blockIdx.x * 128 + wm * 32;
  const int bcol0 = blockIdx.y * 128 + wn * 64;

  v8f acc[2][4];
#pragma unroll
  for (int mt = 0; mt < 2; ++mt)
#pragma unroll
    for (int nt = 0; nt < 4; ++nt) acc[mt][nt] = v8f_zero();

  const int ak = lhi * 8;
  const int bk = lhi * 8;

  const __bf16* arow[2];
#pragma unroll
  for (int mt = 0; mt < 2; ++mt)
    arow[mt] = X + (size_t)(arow0 + mt * 16 + l15) * K;

#if SWA_ASYNC
  // ---- async double-buffered staging of the block's W tile (128 x 32) ----
  __shared__ __align__(16) __bf16 Wl[2][128 * 40];
  const int trow  = threadIdx.x >> 1;       // 0..127 (block-relative W row)
  const int thalf = threadIdx.x & 1;        // covers cols thalf*16 .. +15
  const __bf16* wsrc = W + (size_t)(blockIdx.y * 128 + trow) * K + thalf * 16;
  __bf16* wdst0 = &Wl[0][trow * 40 + thalf * 16];
  __bf16* wdst1 = &Wl[1][trow * 40 + thalf * 16];

  // prologue: k = 0 into buffer 0
  async_b128(wsrc,     wdst0);
  async_b128(wsrc + 8, wdst0 + 8);

  int buf = 0;
  for (int k = 0; k < K; k += 32) {
    if (k + 32 < K) {
      __bf16* nd = buf ? wdst0 : wdst1;
      async_b128(wsrc + k + 32,     nd);
      async_b128(wsrc + k + 32 + 8, nd + 8);
      __builtin_amdgcn_s_wait_asynccnt(2);   // current buffer's loads done
    } else {
      __builtin_amdgcn_s_wait_asynccnt(0);
    }
    __syncthreads();

    v16bf a[2], b[4];
#pragma unroll
    for (int mt = 0; mt < 2; ++mt)
      a[mt] = ld_ab(arow[mt] + k + ak, arow[mt] + k + 16 + ak);
    const __bf16* wl = &Wl[buf][0];
#pragma unroll
    for (int nt = 0; nt < 4; ++nt) {
      const __bf16* br = wl + (wn * 64 + nt * 16 + l15) * 40 + bk;
      b[nt] = ld_ab(br, br + 16);
    }
#pragma unroll
    for (int mt = 0; mt < 2; ++mt)
#pragma unroll
      for (int nt = 0; nt < 4; ++nt)
        acc[mt][nt] = __builtin_amdgcn_wmma_f32_16x16x32_bf16(
            false, a[mt], false, b[nt], (short)0, acc[mt][nt], false, false);

    __syncthreads();   // all waves done reading buf before it is re-filled
    buf ^= 1;
  }
#else
  // ---- fallback: direct global loads for B ----
  const __bf16* brow[4];
#pragma unroll
  for (int nt = 0; nt < 4; ++nt)
    brow[nt] = W + (size_t)(bcol0 + nt * 16 + l15) * K;

  for (int k = 0; k < K; k += 32) {
    v16bf a[2], b[4];
#pragma unroll
    for (int mt = 0; mt < 2; ++mt)
      a[mt] = ld_ab(arow[mt] + k + ak, arow[mt] + k + 16 + ak);
#pragma unroll
    for (int nt = 0; nt < 4; ++nt)
      b[nt] = ld_ab(brow[nt] + k + bk, brow[nt] + k + bk + 16);
#pragma unroll
    for (int mt = 0; mt < 2; ++mt)
#pragma unroll
      for (int nt = 0; nt < 4; ++nt)
        acc[mt][nt] = __builtin_amdgcn_wmma_f32_16x16x32_bf16(
            false, a[mt], false, b[nt], (short)0, acc[mt][nt], false, false);
  }
#endif

  // epilogue: C layout -> lane<16 rows v, lane>=16 rows v+8; col = l15
#pragma unroll
  for (int nt = 0; nt < 4; ++nt) {
    const int col = bcol0 + nt * 16 + l15;
    const float bv = bias[col];
#pragma unroll
    for (int mt = 0; mt < 2; ++mt) {
#pragma unroll
      for (int v = 0; v < 8; ++v) {
        const int row = arow0 + mt * 16 + v + lhi * 8;
        const float val = acc[mt][nt][v] + bv;
        if (OUT_F32)
          ((float*)Yv)[(size_t)row * N + col] = val;
        else
          ((__bf16*)Yv)[(size_t)row * N + col] = (__bf16)val;
      }
    }
  }
}

// --------------------------- window attention ------------------------------
// One block = half of one (b,h,window): 128 query rows, 8 waves x 16 rows.
// Streaming (flash) softmax over 256 keys, 32 columns per step.
#define SWA_WIN 256
#define SWA_DK  64

__global__ __launch_bounds__(256) void swa_attn(
    const __bf16* __restrict__ Q, const __bf16* __restrict__ Km,
    const __bf16* __restrict__ Vm, __bf16* __restrict__ ctx,
    int S, int D, int H) {
  __shared__ __align__(16) __bf16 VT[SWA_DK][SWA_WIN + 8];  // V transposed
  __shared__ __align__(16) __bf16 P[8][16 * 40];            // per-wave P relayout

  const int nW = S / SWA_WIN;
  int bi   = blockIdx.x;
  const int half = bi & 1;            bi >>= 1;
  const int w    = bi % nW;           bi /= nW;
  const int h    = bi % H;
  const int b    = bi / H;

  const int lane = threadIdx.x & 31;
  const int wid  = threadIdx.x >> 5;
  const int l15  = lane & 15, lhi = lane >> 4;

  const size_t base = ((size_t)b * S + (size_t)w * SWA_WIN) * D + (size_t)h * SWA_DK;

  // stage V^T into LDS (coalesced read, one time)
  for (int idx = threadIdx.x; idx < SWA_WIN * SWA_DK; idx += 256) {
    const int kk = idx >> 6, nn = idx & 63;
    VT[nn][kk] = Vm[base + (size_t)kk * D + nn];
  }
  __syncthreads();

  const int r0 = half * 128 + wid * 16;
  const __bf16* qrow = Q + base + (size_t)(r0 + l15) * D;
  v16bf q0 = ld_ab(qrow + lhi * 8,      qrow + 16 + lhi * 8);   // dk 0..31
  v16bf q1 = ld_ab(qrow + 32 + lhi * 8, qrow + 48 + lhi * 8);   // dk 32..63
  // fold 1/sqrt(dk)=0.125 into Q (exact in bf16: exponent shift)
#pragma unroll
  for (int i = 0; i < 16; ++i) {
    q0[i] = (__bf16)((float)q0[i] * 0.125f);
    q1[i] = (__bf16)((float)q1[i] * 0.125f);
  }

  float m[8], l[8];
  v8f o[4];
#pragma unroll
  for (int v = 0; v < 8; ++v) { m[v] = -3.0e38f; l[v] = 0.0f; }
#pragma unroll
  for (int nt = 0; nt < 4; ++nt) o[nt] = v8f_zero();

  __bf16* prow = &P[wid][0];

  for (int c0 = 0; c0 < SWA_WIN; c0 += 32) {
    // ---- S = Q K^T for 32 key columns (2 tiles x 2 k-steps) ----
    v8f s[2];
#pragma unroll
    for (int t = 0; t < 2; ++t) {
      const __bf16* krow = Km + base + (size_t)(c0 + t * 16 + l15) * D;
      const v16bf kb0 = ld_ab(krow + lhi * 8,      krow + 16 + lhi * 8);
      const v16bf kb1 = ld_ab(krow + 32 + lhi * 8, krow + 48 + lhi * 8);
      v8f z = v8f_zero();
      z = __builtin_amdgcn_wmma_f32_16x16x32_bf16(false, q0, false, kb0,
                                                  (short)0, z, false, false);
      z = __builtin_amdgcn_wmma_f32_16x16x32_bf16(false, q1, false, kb1,
                                                  (short)0, z, false, false);
      s[t] = z;
    }

    // ---- online softmax (per accumulator slot = one query row per half) ----
    float p0[8], p1[8];
#pragma unroll
    for (int v = 0; v < 8; ++v) {
      const float a0 = s[0][v];
      const float a1 = s[1][v];
      const float mx = red16_max(fmaxf(a0, a1));
      const float mn = fmaxf(m[v], mx);
      const float f  = __expf(m[v] - mn);
      p0[v] = __expf(a0 - mn);
      p1[v] = __expf(a1 - mn);
      const float rs = red16_sum(p0[v] + p1[v]);
      l[v] = l[v] * f + rs;
      m[v] = mn;
#pragma unroll
      for (int nt = 0; nt < 4; ++nt) o[nt][v] *= f;
    }

    // ---- relayout P (C-layout f32) -> A-layout bf16 via per-wave LDS ----
#pragma unroll
    for (int v = 0; v < 8; ++v) {
      const int row = v + lhi * 8;
      prow[row * 40 + l15]      = (__bf16)p0[v];
      prow[row * 40 + 16 + l15] = (__bf16)p1[v];
    }
    asm volatile("s_wait_dscnt 0x0" ::: "memory");  // per-wave LDS is HW-ordered
    const __bf16* pr = prow + l15 * 40;
    const v16bf pa = ld_ab(pr + lhi * 16, pr + lhi * 16 + 8);

    // ---- O += P V (V columns contiguous in VT) ----
#pragma unroll
    for (int nt = 0; nt < 4; ++nt) {
      const __bf16* vrow = &VT[nt * 16 + l15][c0 + lhi * 16];
      const v16bf vb = ld_ab(vrow, vrow + 8);
      o[nt] = __builtin_amdgcn_wmma_f32_16x16x32_bf16(false, pa, false, vb,
                                                      (short)0, o[nt], false, false);
    }
  }

  // ---- finalize: ctx[b, s, h*64 + n] = O / l ----
#pragma unroll
  for (int v = 0; v < 8; ++v) {
    const float inv = 1.0f / l[v];
    const int row = r0 + v + lhi * 8;
    const size_t cbase = base + (size_t)row * D;
#pragma unroll
    for (int nt = 0; nt < 4; ++nt)
      ctx[cbase + nt * 16 + l15] = (__bf16)(o[nt][v] * inv);
  }
}

// ------------------------------ launch -------------------------------------
extern "C" void kernel_launch(void* const* d_in, const int* in_sizes, int n_in,
                              void* d_out, int out_size, void* d_ws, size_t ws_size,
                              hipStream_t stream) {
  const int B = 2, S = 4096, D = 1024, H = 16;
  const int M = B * S;               // 8192
  const size_t NX = (size_t)M * D;   // 8,388,608 elements
  const size_t NW = (size_t)D * D;   // 1,048,576 elements

  const float* q  = (const float*)d_in[0];
  const float* k  = (const float*)d_in[1];
  const float* v  = (const float*)d_in[2];
  const float* wq = (const float*)d_in[3];
  const float* bq = (const float*)d_in[4];
  const float* wk = (const float*)d_in[5];
  const float* bk = (const float*)d_in[6];
  const float* wv = (const float*)d_in[7];
  const float* bv = (const float*)d_in[8];
  const float* wo = (const float*)d_in[9];
  const float* bo = (const float*)d_in[10];

  char* ws = (char*)d_ws;
  __bf16* Xq  = (__bf16*)(ws);
  __bf16* Xk  = (__bf16*)(ws + NX * 2);
  __bf16* Xv  = (__bf16*)(ws + NX * 4);
  __bf16* Wq  = (__bf16*)(ws + NX * 6);
  __bf16* Wk  = (__bf16*)(ws + NX * 6 + NW * 2);
  __bf16* Wv  = (__bf16*)(ws + NX * 6 + NW * 4);
  __bf16* Wo  = (__bf16*)(ws + NX * 6 + NW * 6);
  __bf16* Qb  = (__bf16*)(ws + NX * 6 + NW * 8);
  __bf16* Kb  = (__bf16*)(ws + NX * 8 + NW * 8);
  __bf16* Vb  = (__bf16*)(ws + NX * 10 + NW * 8);
  __bf16* CTX = Xq;  // X buffers are dead after the QKV GEMMs

  // 1) convert to bf16 (vectorized x4)
  swa_cvt_bf16<<<2048, 256, 0, stream>>>(q,  Xq, (int)(NX / 4));
  swa_cvt_bf16<<<2048, 256, 0, stream>>>(k,  Xk, (int)(NX / 4));
  swa_cvt_bf16<<<2048, 256, 0, stream>>>(v,  Xv, (int)(NX / 4));
  swa_cvt_bf16<<<512, 256, 0, stream>>>(wq, Wq, (int)(NW / 4));
  swa_cvt_bf16<<<512, 256, 0, stream>>>(wk, Wk, (int)(NW / 4));
  swa_cvt_bf16<<<512, 256, 0, stream>>>(wv, Wv, (int)(NW / 4));
  swa_cvt_bf16<<<512, 256, 0, stream>>>(wo, Wo, (int)(NW / 4));

  // 2) Q/K/V projections
  dim3 gg(M / 128, D / 128);  // 64 x 8
  swa_gemm_bf16<false><<<gg, 256, 0, stream>>>(Xq, Wq, bq, Qb, M, D, D);
  swa_gemm_bf16<false><<<gg, 256, 0, stream>>>(Xk, Wk, bk, Kb, M, D, D);
  swa_gemm_bf16<false><<<gg, 256, 0, stream>>>(Xv, Wv, bv, Vb, M, D, D);

  // 3) windowed attention: B*H*nW windows, 2 blocks per window
  const int nW_ = S / SWA_WIN;
  swa_attn<<<B * H * nW_ * 2, 256, 0, stream>>>(Qb, Kb, Vb, CTX, S, D, H);

  // 4) output projection -> f32 d_out
  swa_gemm_bf16<true><<<gg, 256, 0, stream>>>(CTX, Wo, bo, (float*)d_out, M, D, D);
}
